// LSTMAutoencoder_82240033784093
// MI455X (gfx1250) — matrix operand order repaired and checked by reading, not compiled, not run on hardware
//
#include <hip/hip_runtime.h>

// ---------------------------------------------------------------------------
// Types for CDNA5 WMMA (wave32): bf16 16x16x32, f32 accumulate.
// ---------------------------------------------------------------------------
typedef __attribute__((ext_vector_type(16))) __bf16     bf16x16;
typedef __attribute__((ext_vector_type(8)))  float      f32x8;
typedef __attribute__((ext_vector_type(4)))  unsigned   u32x4;

union FragAB { bf16x16 v; u32x4 q[2]; };

// ---------------------------------------------------------------------------
// Activations (f32)
// ---------------------------------------------------------------------------
__device__ __forceinline__ float sigmoid_f(float x) {
    return 1.f / (1.f + __expf(-x));
}
__device__ __forceinline__ float tanh_f(float x) {
    float a = fabsf(x);
    float e = __expf(-2.f * a);
    float t = (1.f - e) / (1.f + e);
    return copysignf(t, x);
}

// ---------------------------------------------------------------------------
// fp32 -> bf16 cast (same layout)
// ---------------------------------------------------------------------------
__global__ void k_cast_bf16(const float* __restrict__ src, __bf16* __restrict__ dst, long n) {
    long i = (long)blockIdx.x * blockDim.x + threadIdx.x;
    if (i < n) dst[i] = (__bf16)src[i];
}

// ---------------------------------------------------------------------------
// fp32 [R][C] -> bf16 [C][R] transpose-cast (for W^T, U^T so WMMA B-fragments
// are contiguous along K)
// ---------------------------------------------------------------------------
__global__ void k_tcast_bf16(const float* __restrict__ src, __bf16* __restrict__ dst,
                             int R, int C) {
    long i = (long)blockIdx.x * blockDim.x + threadIdx.x;
    long n = (long)R * C;
    if (i >= n) return;
    int r = (int)(i / C);
    int c = (int)(i % C);
    dst[(long)c * R + r] = (__bf16)src[i];
}

// ---------------------------------------------------------------------------
// WMMA GEMM: C = A[M,K](bf16 rm) * Bt[N,K](bf16 rm)^T + bias
// Register-blocked 2 M-tiles x 2 N-tiles per wave (32x32 output): 4
// accumulator chains, 8 b128 loads per 4 WMMAs (A reused 2x, B reused 2x).
// __launch_bounds__(256,1): occupancy is irrelevant for this short, massively
// parallel kernel — give the scheduler registers to pipeline loads across
// k-iterations (round-2 showed spills + full loadcnt drains at 2x4 blocking).
// Input rows of A are m = b*T + t ([B,T,K] flattening). If remapT > 0, the
// output row for input row m is (m%T)*B_ + (m/T): C is emitted in [T][B][N]
// order so the LSTM's per-step gate slice is one contiguous, L2-hot block.
// ---------------------------------------------------------------------------
__global__ __launch_bounds__(256, 1)
void k_gemm_bias(const __bf16* __restrict__ A,
                 const __bf16* __restrict__ Bt,
                 const float*  __restrict__ bias,
                 float* __restrict__ C,
                 int M, int N, int K,
                 int remapT, int B_) {
    int wave = (int)((blockIdx.x * (long)blockDim.x + threadIdx.x) >> 5);
    int lane = threadIdx.x & 31;
    int MG = M >> 5;                    // groups of 2 M-tiles
    int NG = N >> 5;                    // groups of 2 N-tiles
    if (wave >= MG * NG) return;        // wave-uniform: EXEC stays all-1
    int gm = wave / NG;
    int gn = wave - gm * NG;
    int m0 = gm * 32;
    int n0 = gn * 32;
    int ln = lane & 15;
    int hi = lane >> 4;

    const __bf16* arow0 = A + (size_t)(m0 + ln) * K;        // A tile 0 rows
    const __bf16* arow1 = arow0 + (size_t)16 * K;           // A tile 1 rows
    const __bf16* brow0 = Bt + (size_t)(n0 + ln) * K;       // B tile 0 cols
    const __bf16* brow1 = brow0 + (size_t)16 * K;           // B tile 1 cols

    float bv0 = bias ? bias[n0 + ln] : 0.f;
    float bv1 = bias ? bias[n0 + 16 + ln] : 0.f;
    f32x8 acc00 = { bv0, bv0, bv0, bv0, bv0, bv0, bv0, bv0 };
    f32x8 acc01 = { bv1, bv1, bv1, bv1, bv1, bv1, bv1, bv1 };
    f32x8 acc10 = acc00;
    f32x8 acc11 = acc01;

    for (int k0 = 0; k0 < K; k0 += 32) {
        FragAB a0, a1, b0, b1;
        a0.q[0] = *(const u32x4*)(arow0 + k0 + hi * 8);
        a0.q[1] = *(const u32x4*)(arow0 + k0 + 16 + hi * 8);
        a1.q[0] = *(const u32x4*)(arow1 + k0 + hi * 8);
        a1.q[1] = *(const u32x4*)(arow1 + k0 + 16 + hi * 8);
        const __bf16* bp0 = brow0 + k0 + hi * 8;
        const __bf16* bp1 = brow1 + k0 + hi * 8;
        b0.q[0] = *(const u32x4*)(bp0);
        b0.q[1] = *(const u32x4*)(bp0 + 16);
        b1.q[0] = *(const u32x4*)(bp1);
        b1.q[1] = *(const u32x4*)(bp1 + 16);
        acc00 = __builtin_amdgcn_wmma_f32_16x16x32_bf16(false, a0.v, false, b0.v,
                                                        (short)0, acc00, false, false);
        acc01 = __builtin_amdgcn_wmma_f32_16x16x32_bf16(false, a0.v, false, b1.v,
                                                        (short)0, acc01, false, false);
        acc10 = __builtin_amdgcn_wmma_f32_16x16x32_bf16(false, a1.v, false, b0.v,
                                                        (short)0, acc10, false, false);
        acc11 = __builtin_amdgcn_wmma_f32_16x16x32_bf16(false, a1.v, false, b1.v,
                                                        (short)0, acc11, false, false);
    }

    // Epilogue. C/D f32 16x16: lane col n = lane%16, rows j + 8*(lane/16).
    f32x8 accs[2][2] = { { acc00, acc01 }, { acc10, acc11 } };
#pragma unroll
    for (int mi = 0; mi < 2; ++mi) {
        int mrow = m0 + mi * 16;                       // input row base (= b*T + t0)
        if (remapT > 0) {
            int bidx = mrow / remapT;
            int t0   = mrow - bidx * remapT;
            float* crow = C + ((size_t)(t0 + hi * 8) * B_ + bidx) * N + n0 + ln;
#pragma unroll
            for (int q = 0; q < 2; ++q)
#pragma unroll
                for (int j = 0; j < 8; ++j)
                    crow[(size_t)j * B_ * N + q * 16] = accs[mi][q][j];
        } else {
            float* crow = C + (size_t)(mrow + hi * 8) * N + n0 + ln;
#pragma unroll
            for (int q = 0; q < 2; ++q)
#pragma unroll
                for (int j = 0; j < 8; ++j)
                    crow[(size_t)j * N + q * 16] = accs[mi][q][j];
        }
    }
}

// ---------------------------------------------------------------------------
// Fused LSTM recurrence for one layer.
// Grid: B/16 blocks (batch tiles), blockDim = 2*H (H/16 waves).
// Wave w owns hidden columns [16w,16w+16) and all four gate tiles for them,
// so the c/h update is wave-local. h_t lives in LDS (bf16 [16][H]); c lives
// in registers (f32x8) across all T steps. xw is in [T][B][4H] order so the
// per-step gate slice is one contiguous, L2-hot 16x4H block.
// ---------------------------------------------------------------------------
#define LSTM_MAXH 512

__global__ void k_lstm(const float*  __restrict__ xw,      // [T, B, 4H]
                       const __bf16* __restrict__ Ut,      // [4H, H] = U^T
                       __bf16* __restrict__ outSeq,        // [B*T, H] bf16 or null
                       float*  __restrict__ outLast,       // [B, H] f32 or null
                       int H, int T, int B, int actTanh) {
    __shared__ __bf16 hbuf[16 * LSTM_MAXH];

    int wave = threadIdx.x >> 5;     // [0, H/16)
    int lane = threadIdx.x & 31;
    int ln   = lane & 15;
    int hi   = lane >> 4;
    int n0   = wave * 16;
    int b0   = blockIdx.x * 16;
    int fourH = 4 * H;

    for (int i = threadIdx.x; i < 16 * H; i += blockDim.x) hbuf[i] = (__bf16)0.f;
    f32x8 c = { 0.f, 0.f, 0.f, 0.f, 0.f, 0.f, 0.f, 0.f };
    __syncthreads();

    for (int t = 0; t < T; ++t) {
        // init accumulators with xW slice (z = xW_t + h U); contiguous block
        const float* xwb = xw + ((size_t)t * B + b0) * fourH;
        f32x8 acc[4];
#pragma unroll
        for (int g = 0; g < 4; ++g) {
            int col = g * H + n0 + ln;
#pragma unroll
            for (int j = 0; j < 8; ++j) {
                int m = j + 8 * hi;
                acc[g][j] = xwb[(size_t)m * fourH + col];
            }
        }
        // z += h @ U  (A from LDS h, B from U^T hot in L2)
        for (int k0 = 0; k0 < H; k0 += 32) {
            FragAB a;
            const __bf16* ar = hbuf + ln * H;
            a.q[0] = *(const u32x4*)(ar + k0 + hi * 8);
            a.q[1] = *(const u32x4*)(ar + k0 + 16 + hi * 8);
#pragma unroll
            for (int g = 0; g < 4; ++g) {
                FragAB b;
                const __bf16* bp = Ut + (size_t)(g * H + n0 + ln) * H + k0 + hi * 8;
                b.q[0] = *(const u32x4*)(bp);
                b.q[1] = *(const u32x4*)(bp + 16);
                acc[g] = __builtin_amdgcn_wmma_f32_16x16x32_bf16(false, a.v, false, b.v,
                                                                 (short)0, acc[g],
                                                                 false, false);
            }
        }
        // gates + state update (all columns for this wave are local)
        f32x8 hv;
#pragma unroll
        for (int j = 0; j < 8; ++j) {
            float ig = sigmoid_f(acc[0][j]);
            float fg = sigmoid_f(acc[1][j]);
            float gg = actTanh ? tanh_f(acc[2][j]) : fmaxf(acc[2][j], 0.f);
            float og = sigmoid_f(acc[3][j]);
            float cc = fg * c[j] + ig * gg;
            c[j] = cc;
            float ca = actTanh ? tanh_f(cc) : fmaxf(cc, 0.f);
            hv[j] = og * ca;
        }
        __syncthreads();                       // all waves done reading h_{t-1}
#pragma unroll
        for (int j = 0; j < 8; ++j) {
            int m = j + 8 * hi;
            hbuf[m * H + n0 + ln] = (__bf16)hv[j];
        }
        __syncthreads();                       // h_t visible to all waves

        if (outSeq) {
#pragma unroll
            for (int j = 0; j < 8; ++j) {
                int m = j + 8 * hi;
                outSeq[((size_t)(b0 + m) * T + t) * H + n0 + ln] = (__bf16)hv[j];
            }
        }
        if (outLast && t == T - 1) {
#pragma unroll
            for (int j = 0; j < 8; ++j) {
                int m = j + 8 * hi;
                outLast[(size_t)(b0 + m) * H + n0 + ln] = hv[j];
            }
        }
    }
}

// ---------------------------------------------------------------------------
// Orchestration
// ---------------------------------------------------------------------------
extern "C" void kernel_launch(void* const* d_in, const int* in_sizes, int n_in,
                              void* d_out, int out_size, void* d_ws, size_t ws_size,
                              hipStream_t stream) {
    (void)in_sizes; (void)n_in; (void)out_size; (void)ws_size;

    const int B = 64, T = 256;
    struct LD { int I, H; };
    const LD dims[6] = { {256,512}, {512,256}, {256,64}, {64,256}, {256,512}, {512,256} };

    char* ws = (char*)d_ws;
    size_t off = 0;
    auto wsa = [&](size_t bytes) -> void* {
        void* p = (void*)(ws + off);
        off += (bytes + 255) & ~(size_t)255;
        return p;
    };

    __bf16* seqA = (__bf16*)wsa((size_t)B * T * 512 * sizeof(__bf16));
    __bf16* seqB = (__bf16*)wsa((size_t)B * T * 512 * sizeof(__bf16));
    float*  xw   = (float*) wsa((size_t)B * T * 2048 * sizeof(float));
    __bf16* Wt   = (__bf16*)wsa((size_t)2048 * 512 * sizeof(__bf16));
    __bf16* Ut   = (__bf16*)wsa((size_t)2048 * 512 * sizeof(__bf16));

    // cast input x [B,T,256] -> bf16
    const float* x = (const float*)d_in[0];
    long nx = (long)B * T * 256;
    k_cast_bf16<<<(unsigned)((nx + 255) / 256), 256, 0, stream>>>(x, seqA, nx);

    __bf16* cur = seqA;
    __bf16* nxt = seqB;

    for (int l = 0; l < 6; ++l) {
        const float* W  = (const float*)d_in[1 + 3 * l];
        const float* U  = (const float*)d_in[2 + 3 * l];
        const float* bb = (const float*)d_in[3 + 3 * l];
        int I = dims[l].I, H = dims[l].H, N4 = 4 * H;

        long nw = (long)I * N4;
        k_tcast_bf16<<<(unsigned)((nw + 255) / 256), 256, 0, stream>>>(W, Wt, I, N4);
        long nu = (long)H * N4;
        k_tcast_bf16<<<(unsigned)((nu + 255) / 256), 256, 0, stream>>>(U, Ut, H, N4);

        int M = B * T;
        long groups = (long)(M / 32) * (N4 / 32);   // 2x2 tile blocks per wave
        k_gemm_bias<<<(unsigned)((groups + 7) / 8), 256, 0, stream>>>(
            cur, Wt, bb, xw, M, N4, I, /*remapT=*/T, /*B_=*/B);

        bool last = (l == 5);
        k_lstm<<<B / 16, 2 * H, 0, stream>>>(xw, Ut,
                                             last ? (__bf16*)nullptr : nxt,
                                             last ? (float*)d_out : (float*)nullptr,
                                             H, T, B, last ? 1 : 0);
        __bf16* tmp = cur; cur = nxt; nxt = tmp;
    }
}
